// APPFPModule_83227876262279
// MI455X (gfx1250) — matrix operand order repaired and checked by reading, not compiled, not run on hardware
//
#include <hip/hip_runtime.h>

// ---------------------------------------------------------------------------
// Problem constants (from reference setup_inputs)
// ---------------------------------------------------------------------------
#define B_    2
#define NPTS  65536          // unknown points per batch
#define MPTS  16384          // known points per batch
#define C1_   128            // unknow_feats channels
#define C2_   256            // known_feats channels
#define NTOT  (B_ * NPTS)    // 131072 columns in the GEMMs
#define KX    (C2_ + C1_)    // 384  (GEMM1 K)
#define H1_   256            // GEMM1 output channels
#define H2_   128            // GEMM2 output channels
#define EPS_  1e-5f

typedef __attribute__((ext_vector_type(16))) __bf16 v16bf;
typedef __attribute__((ext_vector_type(8)))  __bf16 v8bf;
typedef __attribute__((ext_vector_type(8)))  float  v8f;
typedef __attribute__((ext_vector_type(4)))  float  v4f;
typedef int v4i __attribute__((vector_size(4 * sizeof(int))));  // matches builtin param

// ---------------------------------------------------------------------------
// Kernel 1: brute-force 1-NN, LDS-tiled over known points.
// grid = (NPTS/256, B_), block = 256 (8 wave32).
// ---------------------------------------------------------------------------
__global__ __launch_bounds__(256)
void nn_kernel(const float* __restrict__ unknown,
               const float* __restrict__ known,
               int* __restrict__ idx) {
    const int b = blockIdx.y;
    const int p = blockIdx.x * 256 + threadIdx.x;
    const float* up = unknown + ((size_t)b * NPTS + p) * 3;
    const float ux = up[0], uy = up[1], uz = up[2];
    const float* kb = known + (size_t)b * MPTS * 3;

    __shared__ float sk[512 * 3];

    float best = 3.402823466e38f;
    int   bi   = 0;
    for (int t = 0; t < MPTS; t += 512) {
        #pragma unroll
        for (int j = 0; j < 6; ++j)
            sk[threadIdx.x + 256 * j] = kb[t * 3 + threadIdx.x + 256 * j];
        __syncthreads();
        #pragma unroll 8
        for (int j = 0; j < 512; ++j) {
            const float dx = ux - sk[j * 3 + 0];
            const float dy = uy - sk[j * 3 + 1];
            const float dz = uz - sk[j * 3 + 2];
            const float d  = dx * dx + dy * dy + dz * dz;
            if (d < best) { best = d; bi = t + j; }   // strict <, keeps first (matches argmin)
        }
        __syncthreads();
    }
    idx[(size_t)b * NPTS + p] = bi;
}

// ---------------------------------------------------------------------------
// Kernel 2: weight swizzle  f32 (MxK row-major) -> bf16 in per-lane contiguous-k
// WMMA A-fragment order: frag(mTile, ks, lane) = 16 contiguous bf16, where
// vector elem q = W[mTile*16 + lane%16][ks*32 + 16*(lane/16) + q].
// grid = (M*K)/256.
// ---------------------------------------------------------------------------
__global__ __launch_bounds__(256)
void wswizzle_kernel(const float* __restrict__ w, __bf16* __restrict__ dst,
                     int K, int KS) {
    const int e    = blockIdx.x * 256 + threadIdx.x;   // flat output index
    const int q    = e & 15;
    const int lane = (e >> 4) & 31;
    const int rest = e >> 9;                           // mTile*KS + ks
    const int ks   = rest % KS;
    const int mT   = rest / KS;
    const int v = q >> 1, j = q & 1, h = lane >> 4;
    const int m = mT * 16 + (lane & 15);
    const int k = ks * 32 + h * 16 + q;
    dst[e] = (__bf16)w[(size_t)m * K + k];
}

// ---------------------------------------------------------------------------
// Kernel 3: gather known_feats by 1-NN index, concat with unknow_feats,
// emit bf16 X POINT-MAJOR: Xp[p][c], p in [0,NTOT), c in [0,KX).
// Block = 256 threads = 32 points x 8 channel-chunks of 48.
// Writes are contiguous 96B runs per thread; reads are the inherent gather.
// grid = NTOT/32.
// ---------------------------------------------------------------------------
__global__ __launch_bounds__(256)
void gather_concat_kernel(const float* __restrict__ known_feats,
                          const float* __restrict__ unknow_feats,
                          const int* __restrict__ idx,
                          __bf16* __restrict__ Xp) {
    const int t  = threadIdx.x;
    const int p  = blockIdx.x * 32 + (t >> 3);
    const int c0 = (t & 7) * 48;
    const int b  = p >> 16;                 // NPTS == 65536
    const int i  = p & (NPTS - 1);
    const int j  = idx[p];
    const float* kf = known_feats + (size_t)b * C2_ * MPTS + j;
    const float* uf = unknow_feats + (size_t)b * C1_ * NPTS + i;
    __bf16* xo = Xp + (size_t)p * KX + c0;
    #pragma unroll 8
    for (int q = 0; q < 48; ++q) {
        const int c = c0 + q;
        const float x = (c < C2_) ? kf[(size_t)c * MPTS]
                                  : uf[(size_t)(c - C2_) * NPTS];
        xo[q] = (__bf16)x;
    }
}

// ---------------------------------------------------------------------------
// Kernel 4: bf16 WMMA GEMM   Y = W(MxK) * Xp^T, f32 accumulate.
//   - A fragments: direct 32B global loads from pre-swizzled W (L2-resident).
//   - B operand:   point-major LDS tile sX[64][40]; each lane's fragment is
//                  16 contiguous bf16 -> 2x ds_load_b128 (bank-conflict-free).
//   - sX staged via global_load_async_to_lds_b128 (ASYNCcnt path).
// Block 256 = 8 wave32; block tile 128(M) x 64(N); wave tile 16 x 64.
// grid = (NTOT/64, M/128).  bcnLayout!=0 -> store in reference (B,C,n) layout.
// ---------------------------------------------------------------------------
__global__ __launch_bounds__(256)
void wmma_gemm_kernel(const v16bf* __restrict__ Wsw,
                      const __bf16* __restrict__ Xp,
                      float* __restrict__ Y,
                      int M, int K, int KS, int bcnLayout, int Cn) {
    const int tid   = threadIdx.x;
    const int lane  = tid & 31;
    const int wave  = tid >> 5;
    const int nBase = blockIdx.x * 64;
    const int mBase = blockIdx.y * 128;

    __shared__ __attribute__((aligned(16))) __bf16 sX[64][40];  // 80B pitch

    const int rowA   = lane & 15;
    const int kHalfB = (lane >> 4) * 16;

    // cooperative loader mapping: 16B per thread per K-step
    const int pl = tid >> 2;            // 0..63  (point within tile)
    const int ch = (tid & 3) * 8;       // 0,8,16,24 (k offset within 32)
    const __bf16* gX = Xp + (size_t)(nBase + pl) * K + ch;

    const int aBase = ((mBase >> 4) + wave) * KS;   // fragment row base

    v8f acc[4] = {v8f{}, v8f{}, v8f{}, v8f{}};

    for (int ks = 0; ks < KS; ++ks) {
        // ---- stage X tile (64 pts x 32 k) into LDS -----------------------
#if defined(__gfx1250__) && __has_builtin(__builtin_amdgcn_global_load_async_to_lds_b128) && __has_builtin(__builtin_amdgcn_s_wait_asynccnt)
        __builtin_amdgcn_global_load_async_to_lds_b128(
            (__attribute__((address_space(1))) v4i*)(gX + ks * 32),
            (__attribute__((address_space(3))) v4i*)&sX[pl][ch], 0, 0);
        __builtin_amdgcn_s_wait_asynccnt(0);
#else
        *(v8bf*)&sX[pl][ch] = *(const v8bf*)(gX + ks * 32);
#endif
        __syncthreads();

        // ---- A fragment: one aligned 32B global load (L2-hot) ------------
        const v16bf a = Wsw[(size_t)(aBase + ks) * 32 + lane];

        // ---- 4 N-subtiles: contiguous 32B B fragment + WMMA --------------
        #pragma unroll
        for (int nt = 0; nt < 4; ++nt) {
            const v16bf bfr = *(const v16bf*)&sX[nt * 16 + rowA][kHalfB];
            acc[nt] = __builtin_amdgcn_wmma_f32_16x16x32_bf16(
                false, a, false, bfr, (short)0, acc[nt], false, false);
        }
        __syncthreads();
    }

    // ---- store D ---------------------------------------------------------
    const int mo   = mBase + wave * 16 + (lane >> 4) * 8;  // 8 contiguous rows
    #pragma unroll
    for (int nt = 0; nt < 4; ++nt) {
        const int col = nBase + nt * 16 + rowA;
        if (bcnLayout) {
            const size_t base = (size_t)(col >> 16) * Cn + (col & (NPTS - 1));
            #pragma unroll
            for (int r = 0; r < 8; ++r)
                Y[base + (size_t)(mo + r) * NPTS] = acc[nt][r];
        } else {
            // point-major: Y[col][mo..mo+7] contiguous -> 2x b128 stores
            float* yp = Y + (size_t)col * M + mo;
            *(v4f*)(yp + 0) = v4f{acc[nt][0], acc[nt][1], acc[nt][2], acc[nt][3]};
            *(v4f*)(yp + 4) = v4f{acc[nt][4], acc[nt][5], acc[nt][6], acc[nt][7]};
        }
    }
}

// ---------------------------------------------------------------------------
// Kernel 5a/5b: BN statistics over point-major Hpre (NTOT x 256), two stage.
// Stage 1: grid=512 blocks, thread t == channel, 256-point slice, coalesced.
// Stage 2: 1 block x 256 threads -> stats[2c]=mean, stats[2c+1]=rsqrt(var+eps)
// ---------------------------------------------------------------------------
__global__ __launch_bounds__(256)
void bnstats_pm_partial_kernel(const float* __restrict__ Hpre,
                               float* __restrict__ pS, float* __restrict__ pS2) {
    const int c  = threadIdx.x;
    const int p0 = blockIdx.x * 256;
    float s = 0.f, s2 = 0.f;
    for (int pp = 0; pp < 256; ++pp) {
        const float x = Hpre[(size_t)(p0 + pp) * 256 + c];
        s += x; s2 += x * x;
    }
    pS [blockIdx.x * 256 + c] = s;
    pS2[blockIdx.x * 256 + c] = s2;
}

__global__ __launch_bounds__(256)
void bnstats_pm_final_kernel(const float* __restrict__ pS, const float* __restrict__ pS2,
                             float* __restrict__ stats) {
    const int c = threadIdx.x;
    float s = 0.f, s2 = 0.f;
    for (int i = 0; i < 512; ++i) { s += pS[i * 256 + c]; s2 += pS2[i * 256 + c]; }
    const float mean = s * (1.0f / (float)NTOT);
    const float var  = s2 * (1.0f / (float)NTOT) - mean * mean;
    stats[2 * c]     = mean;
    stats[2 * c + 1] = rsqrtf(var + EPS_);
}

// ---------------------------------------------------------------------------
// Kernel 6: BN+ReLU -> bf16, point-major in, point-major out (GEMM2 operand).
// grid = 512 blocks; thread == channel; fully coalesced both sides.
// ---------------------------------------------------------------------------
__global__ __launch_bounds__(256)
void bnrelu_bf16_pm_kernel(const float* __restrict__ Hpre, const float* __restrict__ stats,
                           const float* __restrict__ gamma, const float* __restrict__ beta,
                           __bf16* __restrict__ Hb) {
    const int c = threadIdx.x;
    const float mean = stats[2 * c], istd = stats[2 * c + 1];
    const float g = gamma[c], bb = beta[c];
    const int p0 = blockIdx.x * 256;
    for (int pp = 0; pp < 256; ++pp) {
        const size_t o = (size_t)(p0 + pp) * 256 + c;
        float y = (Hpre[o] - mean) * istd * g + bb;
        Hb[o] = (__bf16)(y > 0.f ? y : 0.f);
    }
}

// ---------------------------------------------------------------------------
// Kernel 7: BN stats over d_out in (B,C,n) layout (contiguous per channel).
// grid = C blocks, block = 256.
// ---------------------------------------------------------------------------
__global__ __launch_bounds__(256)
void bnstats_bcn_kernel(const float* __restrict__ Y, float* __restrict__ stats, int Cn) {
    const int c = blockIdx.x;
    float s = 0.f, s2 = 0.f;
    for (int p = threadIdx.x; p < NTOT; p += 256) {
        const size_t src = (size_t)(p >> 16) * Cn + (size_t)c * NPTS + (p & (NPTS - 1));
        const float x = Y[src];
        s += x; s2 += x * x;
    }
    __shared__ float rs[256], rs2[256];
    rs[threadIdx.x] = s; rs2[threadIdx.x] = s2;
    __syncthreads();
    for (int o = 128; o > 0; o >>= 1) {
        if (threadIdx.x < o) {
            rs[threadIdx.x]  += rs[threadIdx.x + o];
            rs2[threadIdx.x] += rs2[threadIdx.x + o];
        }
        __syncthreads();
    }
    if (threadIdx.x == 0) {
        const float mean = rs[0] * (1.0f / (float)NTOT);
        const float var  = rs2[0] * (1.0f / (float)NTOT) - mean * mean;
        stats[2 * c]     = mean;
        stats[2 * c + 1] = rsqrtf(var + EPS_);
    }
}

// ---------------------------------------------------------------------------
// Kernel 8: BN+ReLU in place on d_out ((B,C,n) layout).
// grid = C * 512.
// ---------------------------------------------------------------------------
__global__ __launch_bounds__(256)
void bnrelu_inplace_bcn_kernel(float* __restrict__ Y, const float* __restrict__ stats,
                               const float* __restrict__ gamma, const float* __restrict__ beta,
                               int Cn) {
    const int c = blockIdx.x >> 9;
    const int p = ((blockIdx.x & 511) << 8) + threadIdx.x;
    const size_t src = (size_t)(p >> 16) * Cn + (size_t)c * NPTS + (p & (NPTS - 1));
    const float mean = stats[2 * c], istd = stats[2 * c + 1];
    const float g = gamma[c], bb = beta[c];
    const float x = Y[src];
    const float y = (x - mean) * istd * g + bb;
    Y[src] = y > 0.f ? y : 0.f;
}

// ---------------------------------------------------------------------------
// Host launcher
// ---------------------------------------------------------------------------
extern "C" void kernel_launch(void* const* d_in, const int* in_sizes, int n_in,
                              void* d_out, int out_size, void* d_ws, size_t ws_size,
                              hipStream_t stream) {
    const float* unknown      = (const float*)d_in[0];
    const float* known        = (const float*)d_in[1];
    const float* unknow_feats = (const float*)d_in[2];
    const float* known_feats  = (const float*)d_in[3];
    const float* w1           = (const float*)d_in[4];
    const float* gamma1       = (const float*)d_in[5];
    const float* beta1        = (const float*)d_in[6];
    const float* w2           = (const float*)d_in[7];
    const float* gamma2       = (const float*)d_in[8];
    const float* beta2        = (const float*)d_in[9];
    float* out = (float*)d_out;

    // Workspace carve-up (all offsets 256-B aligned)
    char* ws = (char*)d_ws;
    int*    idx    = (int*)ws;                                //    524288 B
    __bf16* W1sw   = (__bf16*)(ws + 524288);                  //    196608 B
    __bf16* W2sw   = (__bf16*)(ws + 720896);                  //     65536 B
    float*  stats1 = (float*)(ws + 786432);                   //      2048 B
    float*  stats2 = (float*)(ws + 788480);                   //      2048 B
    float*  pS     = (float*)(ws + 790528);                   //    524288 B
    float*  pS2    = (float*)(ws + 1314816);                  //    524288 B
    __bf16* Xp     = (__bf16*)(ws + 1839104);                 // 100663296 B (NTOT x KX)
    float*  Hpre   = (float*)(ws + 1839104 + 100663296);      // 134217728 B (NTOT x H1, pt-major)
    __bf16* Hb     = Xp;  // alias: X region is dead after GEMM1 reads it

    // 1) 1-NN search
    nn_kernel<<<dim3(NPTS / 256, B_), 256, 0, stream>>>(unknown, known, idx);

    // 2) weights -> bf16, pre-swizzled into WMMA A-fragment order
    wswizzle_kernel<<<(H1_ * KX) / 256, 256, 0, stream>>>(w1, W1sw, KX, KX / 32);
    wswizzle_kernel<<<(H2_ * H1_) / 256, 256, 0, stream>>>(w2, W2sw, H1_, H1_ / 32);

    // 3) gather + concat -> point-major bf16 X
    gather_concat_kernel<<<NTOT / 32, 256, 0, stream>>>(known_feats, unknow_feats, idx, Xp);

    // 4) GEMM1: Hpre(pt-major, NTOT x 256) = W1 * X
    wmma_gemm_kernel<<<dim3(NTOT / 64, H1_ / 128), 256, 0, stream>>>(
        (const v16bf*)W1sw, Xp, Hpre, H1_, KX, KX / 32, /*bcn=*/0, /*Cn=*/0);

    // 5) BN1 stats (two-stage, deterministic) + apply -> bf16 H (pt-major)
    bnstats_pm_partial_kernel<<<512, 256, 0, stream>>>(Hpre, pS, pS2);
    bnstats_pm_final_kernel<<<1, 256, 0, stream>>>(pS, pS2, stats1);
    bnrelu_bf16_pm_kernel<<<512, 256, 0, stream>>>(Hpre, stats1, gamma1, beta1, Hb);

    // 6) GEMM2: out = W2 * H, stored directly in (B, C, n) layout
    wmma_gemm_kernel<<<dim3(NTOT / 64, H2_ / 128), 256, 0, stream>>>(
        (const v16bf*)W2sw, Hb, out, H2_, H1_, H1_ / 32, /*bcn=*/1, /*Cn=*/H2_ * NPTS);

    // 7) BN2 stats + apply in place on d_out
    bnstats_bcn_kernel<<<H2_, 256, 0, stream>>>(out, stats2, H2_ * NPTS);
    bnrelu_inplace_bcn_kernel<<<H2_ * 512, 256, 0, stream>>>(
        out, stats2, gamma2, beta2, H2_ * NPTS);
}
